// CogView4TransformerBlock_15049565405323
// MI455X (gfx1250) — matrix operand order repaired and checked by reading, not compiled, not run on hardware
//
// CogView4 transformer block for MI455X (gfx1250), bf16 WMMA pipeline.
// Compute-bound (~365 GFLOP vs ~16us of HBM traffic) -> all GEMMs + attention
// run on v_wmma_f32_16x16x32_bf16. LDS tiles are staged with the CDNA5 async
// path (global_load_async_to_lds_b128 + s_wait_asynccnt) so tile DMA overlaps
// WMMA without consuming staging VGPRs. wave32, 16x16 tiles per ISA 7.12.2.
#include <hip/hip_runtime.h>
#include <hip/hip_bf16.h>
#include <math.h>

typedef __bf16 bf16;
typedef __bf16 v16bf __attribute__((ext_vector_type(16)));
typedef float  v8f   __attribute__((ext_vector_type(8)));

#define WMMA_B16(a, b, c) \
  __builtin_amdgcn_wmma_f32_16x16x32_bf16(false, (a), false, (b), (short)0, (c), false, false)

#define D_MODEL 2560
#define SEQ_T   128
#define SEQ_I   1920
#define SEQ_S   2048
#define N_TE    512
#define N_FF    10240
#define N_HEADS 64
#define HDIM    40
#define HPAD    64

static __device__ __forceinline__ v8f vzero() {
  v8f z;
#pragma unroll
  for (int i = 0; i < 8; i++) z[i] = 0.0f;
  return z;
}

// Async 16B copy global -> LDS (per lane). VDST vgpr holds the LDS byte offset
// (= low 32 bits of a generic shared pointer), VADDR the 64-bit global address.
// Tracked by ASYNCcnt (ISA 15.18.3 op 98).
static __device__ __forceinline__ void async_ld_b128(void* lds, const void* g) {
  unsigned loff = (unsigned)(size_t)lds;
  unsigned long long ga = (unsigned long long)(size_t)g;
  asm volatile("global_load_async_to_lds_b128 %0, %1, off"
               :: "v"(loff), "v"(ga)
               : "memory");
}
#define WAIT_ASYNC_4() asm volatile("s_wait_asynccnt 0x4" ::: "memory")
#define WAIT_ASYNC_0() asm volatile("s_wait_asynccnt 0x0" ::: "memory")

// A fragment (16x32, M x K): lane row = lane&15; low lane half holds K 0..7 &
// 16..23, high half K 8..15 & 24..31 (ISA 7.12.2, 16-bit A 16x32).
static __device__ __forceinline__ v16bf frag_a(const bf16* base, int row, int stride,
                                               int k0, int lane) {
  const int hi = (lane >> 4) & 1;
  const bf16* p = base + row * stride + k0;
  union { v16bf v; uint4 q[2]; } u;
  u.q[0] = *(const uint4*)(p + (hi ? 8 : 0));
  u.q[1] = *(const uint4*)(p + (hi ? 24 : 16));
  return u.v;
}

// B fragment (32x16, K x N), source stored N-major / K-contiguous: lane col =
// lane&15, low half K 0..7 & 16..23, high half K 8..15 & 24..31 (matches A).
static __device__ __forceinline__ v16bf frag_b(const bf16* base, int col, int stride,
                                               int k0, int lane) {
  const int hi = (lane >> 4) & 1;
  const bf16* p = base + col * stride + k0;
  union { v16bf v; uint4 q[2]; } u;
  u.q[0] = *(const uint4*)(p + (hi ? 8 : 0));
  u.q[1] = *(const uint4*)(p + (hi ? 24 : 16));
  return u.v;
}

static __device__ __forceinline__ float gelu_tanh(float x) {
  float x3 = x * x * x;
  return 0.5f * x * (1.0f + tanhf(0.7978845608028654f * (x + 0.044715f * x3)));
}

// ---------------------------------------------------------------------------
// adaLN GEMV: emb[30720] = temb[512] @ adaln_w[512,30720] + b
// ---------------------------------------------------------------------------
__global__ __launch_bounds__(256) void adaln_gemv(const float* __restrict__ temb,
                                                  const float* __restrict__ W,
                                                  const float* __restrict__ b,
                                                  float* __restrict__ emb) {
  int n = blockIdx.x * 256 + threadIdx.x;  // < 30720
  float acc = 0.0f;
  for (int k = 0; k < N_TE; k++) acc += temb[k] * W[(size_t)k * (12 * D_MODEL) + n];
  emb[n] = acc + b[n];
}

__global__ __launch_bounds__(256) void build_maskf(const int* __restrict__ am,
                                                   float* __restrict__ mf) {
  int s = blockIdx.x * 256 + threadIdx.x;
  if (s < SEQ_S) mf[s] = (s < SEQ_T) ? (float)am[s] : 1.0f;
}

// ---------------------------------------------------------------------------
// fp32 [R][C] -> bf16 transposed [C][R]  (weights -> Bt layout for WMMA)
// ---------------------------------------------------------------------------
__global__ __launch_bounds__(256) void cvt_t(const float* __restrict__ src,
                                             bf16* __restrict__ dst, int R, int C) {
  __shared__ float t[32][33];
  const int r0 = blockIdx.y * 32, c0 = blockIdx.x * 32;
  const int tx = threadIdx.x & 31, ty = threadIdx.x >> 5;
#pragma unroll
  for (int i = 0; i < 4; i++) {
    int r = ty + i * 8;
    t[r][tx] = src[(size_t)(r0 + r) * C + c0 + tx];
  }
  __syncthreads();
#pragma unroll
  for (int i = 0; i < 4; i++) {
    int c = ty + i * 8;
    dst[(size_t)(c0 + c) * R + r0 + tx] = (bf16)t[tx][c];
  }
}

// ---------------------------------------------------------------------------
// LayerNorm + adaLN modulation -> bf16 row (builds GEMM A matrix)
// rows [0,T) come from enc, rows [T,S) from hid.
// ---------------------------------------------------------------------------
__global__ __launch_bounds__(256) void ln_mod(const float* __restrict__ enc,
                                              const float* __restrict__ hid,
                                              const float* __restrict__ emb,
                                              int sc_enc, int sh_enc, int sc_hid,
                                              int sh_hid, bf16* __restrict__ xout) {
  const int s = blockIdx.x, tid = threadIdx.x;
  const int lane = tid & 31, wid = tid >> 5;
  const float* src = (s < SEQ_T) ? enc + (size_t)s * D_MODEL
                                 : hid + (size_t)(s - SEQ_T) * D_MODEL;
  const float* scv = emb + (size_t)((s < SEQ_T) ? sc_enc : sc_hid) * D_MODEL;
  const float* shv = emb + (size_t)((s < SEQ_T) ? sh_enc : sh_hid) * D_MODEL;
  float xl[10], sum = 0.0f, ss = 0.0f;
#pragma unroll
  for (int i = 0; i < 10; i++) {
    xl[i] = src[i * 256 + tid];
    sum += xl[i];
    ss += xl[i] * xl[i];
  }
  __shared__ float red[2][8];
#pragma unroll
  for (int off = 16; off > 0; off >>= 1) {
    sum += __shfl_xor(sum, off, 32);
    ss += __shfl_xor(ss, off, 32);
  }
  if (lane == 0) { red[0][wid] = sum; red[1][wid] = ss; }
  __syncthreads();
  float ts = 0.0f, tq = 0.0f;
#pragma unroll
  for (int w = 0; w < 8; w++) { ts += red[0][w]; tq += red[1][w]; }
  const float mu = ts / (float)D_MODEL;
  const float var = tq / (float)D_MODEL - mu * mu;
  const float inv = rsqrtf(var + 1e-5f);
#pragma unroll
  for (int i = 0; i < 10; i++) {
    int n = i * 256 + tid;
    float y = (xl[i] - mu) * inv;
    xout[(size_t)s * D_MODEL + n] = (bf16)(y * (1.0f + scv[n]) + shv[n]);
  }
}

// ---------------------------------------------------------------------------
// WMMA GEMM: C[M,N] = A[M,K](bf16) @ Bt[N,K](bf16)^T + bias
// 128x128x32 block tile, 8 waves, wave tile 32x64 (2x4 16x16 accums).
// Tiles staged by async global->LDS DMA, double buffered:
//   issue(kt+1) -> s_wait_asynccnt 4 (tile kt resident) -> barrier -> WMMA.
// EPI=0: fp32 out; EPI=1: gelu -> bf16 out.
// ---------------------------------------------------------------------------
template <int EPI>
__global__ __launch_bounds__(256) void gemm_bf16(const bf16* __restrict__ A,
                                                 const bf16* __restrict__ Bt,
                                                 const float* __restrict__ bias,
                                                 void* __restrict__ Cout, int M, int N,
                                                 int K) {
  const int tid = threadIdx.x, lane = tid & 31, wid = tid >> 5;
  const int ln = lane & 15, hi = (lane >> 4) & 1;
  const int bm = blockIdx.y * 128, bn = blockIdx.x * 128;
  __shared__ __align__(16) bf16 as[2][128 * 40];
  __shared__ __align__(16) bf16 bs[2][128 * 40];
  const int wm = (wid >> 1) * 32;  // 0,32,64,96
  const int wn = (wid & 1) * 64;   // 0,64

  v8f acc[2][4];
#pragma unroll
  for (int i = 0; i < 2; i++)
#pragma unroll
    for (int j = 0; j < 4; j++) acc[i][j] = vzero();

  const bf16* ag = A + (size_t)bm * K;
  const bf16* bg = Bt + (size_t)bn * K;

  // 4 async-load instructions per wave per tile (2 chunks A + 2 chunks B).
  auto issue_tile = [&](int kt, int buf) {
#pragma unroll
    for (int p = 0; p < 2; p++) {
      int c = tid + p * 256;     // 0..511 chunks of 8 bf16
      int r = c >> 2;            // 0..127
      int kc = (c & 3) * 8;      // 0,8,16,24
      async_ld_b128(&as[buf][r * 40 + kc], ag + (size_t)r * K + (size_t)kt * 32 + kc);
      async_ld_b128(&bs[buf][r * 40 + kc], bg + (size_t)r * K + (size_t)kt * 32 + kc);
    }
  };

  const int KT = K / 32;
  issue_tile(0, 0);
  for (int kt = 0; kt < KT; kt++) {
    const int buf = kt & 1;
    if (kt + 1 < KT) {
      issue_tile(kt + 1, buf ^ 1);
      WAIT_ASYNC_4();            // only tile kt+1's 4 loads may remain
    } else {
      WAIT_ASYNC_0();
    }
    __syncthreads();             // tile kt visible to all waves

    v16bf af[2], bfr[4];
#pragma unroll
    for (int i = 0; i < 2; i++) af[i] = frag_a(&as[buf][0], wm + i * 16 + ln, 40, 0, lane);
#pragma unroll
    for (int j = 0; j < 4; j++) bfr[j] = frag_b(&bs[buf][0], wn + j * 16 + ln, 40, 0, lane);
#pragma unroll
    for (int i = 0; i < 2; i++)
#pragma unroll
      for (int j = 0; j < 4; j++) acc[i][j] = WMMA_B16(af[i], bfr[j], acc[i][j]);
    __syncthreads();             // compute done before next iter's DMA lands here
  }

#pragma unroll
  for (int j = 0; j < 4; j++) {
    const int col = bn + wn + j * 16 + ln;
    const float bvv = bias ? bias[col] : 0.0f;
#pragma unroll
    for (int i = 0; i < 2; i++) {
      const int rbase = bm + wm + i * 16 + hi * 8;
#pragma unroll
      for (int v = 0; v < 8; v++) {
        float val = acc[i][j][v] + bvv;
        const size_t idx = (size_t)(rbase + v) * N + col;
        if (EPI == 0) ((float*)Cout)[idx] = val;
        else          ((bf16*)Cout)[idx] = (bf16)gelu_tanh(val);
      }
    }
  }
}

// ---------------------------------------------------------------------------
// Head pack: qkv fp32 [S,D] -> per-head bf16. q,k: LN over 40 + RoPE (rows>=T),
// padded to 64: qp/kp [h][s][64]. v: transposed vt [h][64][s] (B-frag for PV).
// One wave per (s,h); lane covers dims {lane, lane+32}.
// ---------------------------------------------------------------------------
__global__ __launch_bounds__(256) void pack_heads(
    const float* __restrict__ q, const float* __restrict__ k,
    const float* __restrict__ v, const float* __restrict__ cosp,
    const float* __restrict__ sinp, bf16* __restrict__ qp, bf16* __restrict__ kp,
    bf16* __restrict__ vt) {
  const int tid = threadIdx.x, lane = tid & 31, wid = tid >> 5;
  const int p = blockIdx.x * 8 + wid;
  const int s = p >> 6, h = p & 63;
  __shared__ float ybuf[8][40];

#pragma unroll
  for (int which = 0; which < 2; which++) {
    const float* src = which ? k : q;
    bf16* dst = which ? kp : qp;
    const size_t base = (size_t)s * D_MODEL + h * HDIM;
    float x0 = src[base + lane];
    float x1 = (lane < 8) ? src[base + lane + 32] : 0.0f;
    float sum = x0 + x1, ss = x0 * x0 + x1 * x1;
#pragma unroll
    for (int off = 16; off > 0; off >>= 1) {
      sum += __shfl_xor(sum, off, 32);
      ss += __shfl_xor(ss, off, 32);
    }
    const float mu = sum / (float)HDIM;
    const float inv = rsqrtf(ss / (float)HDIM - mu * mu + 1e-5f);
    float y0 = (x0 - mu) * inv, y1 = (x1 - mu) * inv;
    ybuf[wid][lane] = y0;
    if (lane < 8) ybuf[wid][lane + 32] = y1;
    __syncthreads();
    float o0 = y0, o1 = y1;
    if (s >= SEQ_T) {  // RoPE on image tokens
      const int i = s - SEQ_T;
      const int d0 = lane;
      float r0 = (d0 < 20) ? -ybuf[wid][d0 + 20] : ybuf[wid][d0 - 20];
      o0 = y0 * cosp[(size_t)i * HDIM + d0] + r0 * sinp[(size_t)i * HDIM + d0];
      if (lane < 8) {
        const int d1 = lane + 32;  // >= 20
        float r1 = ybuf[wid][d1 - 20];
        o1 = y1 * cosp[(size_t)i * HDIM + d1] + r1 * sinp[(size_t)i * HDIM + d1];
      }
    }
    bf16* drow = dst + ((size_t)h * SEQ_S + s) * HPAD;
    drow[lane] = (bf16)o0;
    drow[lane + 32] = (lane < 8) ? (bf16)o1 : (bf16)0.0f;
    __syncthreads();
  }
  // V: transposed, padded
  const size_t vbase = (size_t)s * D_MODEL + h * HDIM;
  float v0 = v[vbase + lane];
  float v1 = (lane < 8) ? v[vbase + lane + 32] : 0.0f;
  vt[((size_t)h * HPAD + lane) * SEQ_S + s] = (bf16)v0;
  vt[((size_t)h * HPAD + lane + 32) * SEQ_S + s] = (lane < 8) ? (bf16)v1 : (bf16)0.0f;
}

// ---------------------------------------------------------------------------
// Flash attention, one block per (head, 128-row q tile). 8 waves, wave owns 16
// q rows, full 64 (padded) head dims. Online softmax; QK^T and PV via WMMA.
// K/V tiles double-buffered via async global->LDS DMA.
// Reference adds (mq*mk>0)?1:0 to scaled scores (not -inf masking).
// ---------------------------------------------------------------------------
__global__ __launch_bounds__(256) void flash_attn(const bf16* __restrict__ qp,
                                                  const bf16* __restrict__ kp,
                                                  const bf16* __restrict__ vt,
                                                  const float* __restrict__ maskf,
                                                  bf16* __restrict__ outb) {
  const int h = blockIdx.x, qt = blockIdx.y;
  const int tid = threadIdx.x, lane = tid & 31, wid = tid >> 5;
  const int ln = lane & 15, hi = (lane >> 4) & 1;
  __shared__ __align__(16) bf16 qs[128 * 72];
  __shared__ __align__(16) bf16 ps[128 * 72];
  __shared__ __align__(16) bf16 ks[2][64 * 72];
  __shared__ __align__(16) bf16 vs[2][64 * 72];

  // Q tile: 4 async loads per wave
  const bf16* qg = qp + ((size_t)h * SEQ_S + qt * 128) * HPAD;
#pragma unroll
  for (int p = 0; p < 4; p++) {
    int c = tid + p * 256, r = c >> 3, kc = (c & 7) * 8;
    async_ld_b128(&qs[r * 72 + kc], qg + r * HPAD + kc);
  }

  // K/V tile: 4 async loads per wave per tile
  auto issue_kv = [&](int kt, int buf) {
    const bf16* kg = kp + ((size_t)h * SEQ_S + kt * 64) * HPAD;
    const bf16* vg = vt + (size_t)h * HPAD * SEQ_S + kt * 64;
#pragma unroll
    for (int p = 0; p < 2; p++) {
      int c = tid + p * 256, r = c >> 3, kc = (c & 7) * 8;
      async_ld_b128(&ks[buf][r * 72 + kc], kg + r * HPAD + kc);
      async_ld_b128(&vs[buf][r * 72 + kc], vg + (size_t)r * SEQ_S + kc);
    }
  };

  v8f o[4];
#pragma unroll
  for (int t = 0; t < 4; t++) o[t] = vzero();
  float mrow[8], lrow[8], mq[8];
#pragma unroll
  for (int v = 0; v < 8; v++) {
    mrow[v] = -3.0e38f;
    lrow[v] = 0.0f;
    int row = qt * 128 + wid * 16 + hi * 8 + v;
    mq[v] = (qt == 0) ? maskf[row] : 1.0f;
  }
  const float scale = 0.15811388300841897f;  // 40^-0.5
  const int NT = SEQ_S / 64;

  issue_kv(0, 0);
  for (int kt = 0; kt < NT; kt++) {
    const int buf = kt & 1;
    if (kt + 1 < NT) {
      issue_kv(kt + 1, buf ^ 1);
      WAIT_ASYNC_4();            // Q + tile kt resident; only kt+1 outstanding
    } else {
      WAIT_ASYNC_0();
    }
    __syncthreads();

    v8f sa[4];
#pragma unroll
    for (int t = 0; t < 4; t++) sa[t] = vzero();
#pragma unroll
    for (int ds = 0; ds < 2; ds++) {
      v16bf aq = frag_a(qs, wid * 16 + ln, 72, ds * 32, lane);
#pragma unroll
      for (int t = 0; t < 4; t++) {
        v16bf bk = frag_b(&ks[buf][0], t * 16 + ln, 72, ds * 32, lane);
        sa[t] = WMMA_B16(aq, bk, sa[t]);
      }
    }

    float mk[4];
#pragma unroll
    for (int t = 0; t < 4; t++) mk[t] = maskf[kt * 64 + t * 16 + ln];

#pragma unroll
    for (int v = 0; v < 8; v++) {
      float pv[4], rm = -3.0e38f;
#pragma unroll
      for (int t = 0; t < 4; t++) {
        float x = sa[t][v] * scale + ((mq[v] * mk[t] > 0.0f) ? 1.0f : 0.0f);
        pv[t] = x;
        rm = fmaxf(rm, x);
      }
#pragma unroll
      for (int off = 1; off < 16; off <<= 1) rm = fmaxf(rm, __shfl_xor(rm, off, 32));
      const float mn = fmaxf(mrow[v], rm);
      const float corr = __expf(mrow[v] - mn);
      float ls = 0.0f;
#pragma unroll
      for (int t = 0; t < 4; t++) {
        pv[t] = __expf(pv[t] - mn);
        ls += pv[t];
      }
#pragma unroll
      for (int off = 1; off < 16; off <<= 1) ls += __shfl_xor(ls, off, 32);
      lrow[v] = lrow[v] * corr + ls;
      mrow[v] = mn;
#pragma unroll
      for (int t = 0; t < 4; t++) o[t][v] *= corr;
      const int prow = wid * 16 + hi * 8 + v;
#pragma unroll
      for (int t = 0; t < 4; t++) ps[prow * 72 + t * 16 + ln] = (bf16)pv[t];
    }

    // PV: each wave reads only the P rows it wrote (intra-wave LDS dependency).
#pragma unroll
    for (int ds = 0; ds < 2; ds++) {
      v16bf ap = frag_a(ps, wid * 16 + ln, 72, ds * 32, lane);
#pragma unroll
      for (int t = 0; t < 4; t++) {
        v16bf bv = frag_b(&vs[buf][0], t * 16 + ln, 72, ds * 32, lane);
        o[t] = WMMA_B16(ap, bv, o[t]);
      }
    }
    __syncthreads();             // done with buf before next DMA overwrites it
  }

#pragma unroll
  for (int t = 0; t < 4; t++) {
    const int d = t * 16 + ln;
    if (d < HDIM) {
#pragma unroll
      for (int v = 0; v < 8; v++) {
        const int row = qt * 128 + wid * 16 + hi * 8 + v;
        outb[(size_t)row * D_MODEL + h * HDIM + d] = (bf16)(o[t][v] / lrow[v]);
      }
    }
  }
}

// ---------------------------------------------------------------------------
// Gated residual epilogues
// ---------------------------------------------------------------------------
__global__ __launch_bounds__(256) void o_epilogue(const float* __restrict__ o,
                                                  const float* __restrict__ enc_in,
                                                  const float* __restrict__ hid_in,
                                                  const float* __restrict__ emb,
                                                  float* __restrict__ out) {
  const int s = blockIdx.x, tid = threadIdx.x;
#pragma unroll
  for (int i = 0; i < 10; i++) {
    const int n = i * 256 + tid;
    const float ov = o[(size_t)s * D_MODEL + n];
    if (s < SEQ_T) {
      out[(size_t)SEQ_I * D_MODEL + (size_t)s * D_MODEL + n] =
          enc_in[(size_t)s * D_MODEL + n] + ov * emb[5 * D_MODEL + n];
    } else {
      out[(size_t)(s - SEQ_T) * D_MODEL + n] =
          hid_in[(size_t)(s - SEQ_T) * D_MODEL + n] + ov * emb[4 * D_MODEL + n];
    }
  }
}

__global__ __launch_bounds__(256) void ff_epilogue(const float* __restrict__ ff,
                                                   const float* __restrict__ emb,
                                                   float* __restrict__ out) {
  const int s = blockIdx.x, tid = threadIdx.x;
#pragma unroll
  for (int i = 0; i < 10; i++) {
    const int n = i * 256 + tid;
    const float fv = ff[(size_t)s * D_MODEL + n];
    if (s < SEQ_T)
      out[(size_t)SEQ_I * D_MODEL + (size_t)s * D_MODEL + n] += fv * emb[11 * D_MODEL + n];
    else
      out[(size_t)(s - SEQ_T) * D_MODEL + n] += fv * emb[10 * D_MODEL + n];
  }
}

// ---------------------------------------------------------------------------
// Workspace layout (bytes), total ~229 MB with phase reuse.
// ---------------------------------------------------------------------------
static constexpr size_t OFF_W   = 0;                       // 104,857,600 weight bf16 arena
static constexpr size_t OFF_A   = 104857600;               //  62,914,560 fp32/act arena
static constexpr size_t OFF_QP  = OFF_A + 62914560;        //  16,777,216
static constexpr size_t OFF_KP  = OFF_QP + 16777216;
static constexpr size_t OFF_VT  = OFF_KP + 16777216;
static constexpr size_t OFF_X   = OFF_VT + 16777216;       //  10,485,760 bf16 activations
static constexpr size_t OFF_EMB = OFF_X + 10485760;        //     122,880
static constexpr size_t OFF_MF  = OFF_EMB + 122880;        //       8,192

extern "C" void kernel_launch(void* const* d_in, const int* in_sizes, int n_in,
                              void* d_out, int out_size, void* d_ws, size_t ws_size,
                              hipStream_t stream) {
  (void)in_sizes; (void)n_in; (void)out_size; (void)ws_size;
  const float* hid  = (const float*)d_in[0];
  const float* enc  = (const float*)d_in[1];
  const float* temb = (const float*)d_in[2];
  const float* rcos = (const float*)d_in[3];
  const float* rsin = (const float*)d_in[4];
  const int*   am   = (const int*)d_in[5];
  const float* adw  = (const float*)d_in[6];
  const float* adb  = (const float*)d_in[7];
  const float* wq = (const float*)d_in[8];  const float* bq = (const float*)d_in[9];
  const float* wk = (const float*)d_in[10]; const float* bk = (const float*)d_in[11];
  const float* wv = (const float*)d_in[12]; const float* bv = (const float*)d_in[13];
  const float* wo = (const float*)d_in[14]; const float* bo = (const float*)d_in[15];
  const float* w1 = (const float*)d_in[16]; const float* b1 = (const float*)d_in[17];
  const float* w2 = (const float*)d_in[18]; const float* b2 = (const float*)d_in[19];
  float* out = (float*)d_out;
  char* ws = (char*)d_ws;

  bf16* wt0 = (bf16*)(ws + OFF_W);                    // phase weight slot 0
  bf16* wt1 = (bf16*)(ws + OFF_W + 13107200);         // wk_t
  bf16* wt2 = (bf16*)(ws + OFF_W + 26214400);         // wv_t
  bf16* w2t = (bf16*)(ws + OFF_W + 52428800);         // ff_w2^T
  float* qf = (float*)(ws + OFF_A);
  float* kf = (float*)(ws + OFF_A + 20971520);
  float* vf = (float*)(ws + OFF_A + 41943040);
  float* of  = (float*)(ws + OFF_A);                  // o-proj fp32 (reuse)
  bf16*  ff1 = (bf16*)(ws + OFF_A);                   // gelu(ffn1) bf16 (reuse)
  float* ff2 = (float*)(ws + OFF_A + 41943040);       // ffn2 fp32
  bf16* qp = (bf16*)(ws + OFF_QP);
  bf16* kp = (bf16*)(ws + OFF_KP);
  bf16* vt = (bf16*)(ws + OFF_VT);
  bf16* x  = (bf16*)(ws + OFF_X);
  float* emb = (float*)(ws + OFF_EMB);
  float* mf  = (float*)(ws + OFF_MF);

  // 1. adaLN + mask
  adaln_gemv<<<120, 256, 0, stream>>>(temb, adw, adb, emb);
  build_maskf<<<8, 256, 0, stream>>>(am, mf);

  // 2. stage-1 LN + modulate -> x (bf16)
  ln_mod<<<SEQ_S, 256, 0, stream>>>(enc, hid, emb, 3, 1, 2, 0, x);

  // 3. QKV weights -> bf16^T, then WMMA GEMMs
  cvt_t<<<dim3(80, 80), 256, 0, stream>>>(wq, wt0, D_MODEL, D_MODEL);
  cvt_t<<<dim3(80, 80), 256, 0, stream>>>(wk, wt1, D_MODEL, D_MODEL);
  cvt_t<<<dim3(80, 80), 256, 0, stream>>>(wv, wt2, D_MODEL, D_MODEL);
  gemm_bf16<0><<<dim3(20, 16), 256, 0, stream>>>(x, wt0, bq, qf, SEQ_S, D_MODEL, D_MODEL);
  gemm_bf16<0><<<dim3(20, 16), 256, 0, stream>>>(x, wt1, bk, kf, SEQ_S, D_MODEL, D_MODEL);
  gemm_bf16<0><<<dim3(20, 16), 256, 0, stream>>>(x, wt2, bv, vf, SEQ_S, D_MODEL, D_MODEL);

  // 4. head pack (QK-LN + RoPE, V transpose)
  pack_heads<<<SEQ_S * N_HEADS / 8, 256, 0, stream>>>(qf, kf, vf, rcos, rsin, qp, kp, vt);

  // 5. flash attention -> bf16 [S,D] into x-buffer
  flash_attn<<<dim3(N_HEADS, SEQ_S / 128), 256, 0, stream>>>(qp, kp, vt, mf, x);

  // 6. O projection + gated residual into d_out
  cvt_t<<<dim3(80, 80), 256, 0, stream>>>(wo, wt0, D_MODEL, D_MODEL);
  gemm_bf16<0><<<dim3(20, 16), 256, 0, stream>>>(x, wt0, bo, of, SEQ_S, D_MODEL, D_MODEL);
  o_epilogue<<<SEQ_S, 256, 0, stream>>>(of, enc, hid, emb, out);

  // 7. stage-2 LN + modulate (reads residual stream in d_out)
  ln_mod<<<SEQ_S, 256, 0, stream>>>(out + (size_t)SEQ_I * D_MODEL, out, emb, 9, 7, 8, 6, x);

  // 8. FFN: gelu(x @ w1 + b1) @ w2 + b2, then gated residual
  cvt_t<<<dim3(320, 80), 256, 0, stream>>>(w1, wt0, D_MODEL, N_FF);   // w1^T [FF][D]
  cvt_t<<<dim3(80, 320), 256, 0, stream>>>(w2, w2t, N_FF, D_MODEL);   // w2^T [D][FF]
  gemm_bf16<1><<<dim3(80, 16), 256, 0, stream>>>(x, wt0, b1, ff1, SEQ_S, N_FF, D_MODEL);
  gemm_bf16<0><<<dim3(20, 16), 256, 0, stream>>>(ff1, w2t, b2, ff2, SEQ_S, D_MODEL, N_FF);
  ff_epilogue<<<SEQ_S, 256, 0, stream>>>(ff2, emb, out);
}
